// NodeModel_17806934409781
// MI455X (gfx1250) — matrix operand order repaired and checked by reading, not compiled, not run on hardware
//
#include <hip/hip_runtime.h>

typedef __attribute__((ext_vector_type(16))) _Float16 v16h;
typedef __attribute__((ext_vector_type(8)))  _Float16 v8h;
typedef __attribute__((ext_vector_type(8)))  float    v8f;
typedef __attribute__((ext_vector_type(4)))  float    v4f;

#define NNODES 50000
#define NEDGES 500000
#define DIM    256          // node/edge/out dim; in_dim = 2*DIM = 512
#define WBYTES 262144       // 512*256 f16 swizzled weights

// ---------------------------------------------------------------------------
// Convert W [512,256] f32 row-major into f16 B-fragment layout, split into two
// 16B-per-lane half-regions per fragment to cut LDS bank conflicts to 2-way.
// Fragment (kk,nt): K = kk*32.., N = nt*16.. ; per ISA B layout lanes 0-15
// hold K=0..7,16..23 (halves 0..15), lanes 16-31 hold K=8..15,24..31; N = lane&15.
// Linear idx (halves): (kk*16+nt)*512 + r*256 + lane*8 + q, half = r*8+q.
// ---------------------------------------------------------------------------
__global__ void k_swizzle(const float* __restrict__ W, _Float16* __restrict__ Wz) {
  int idx  = blockIdx.x * blockDim.x + threadIdx.x;   // 131072
  int q    = idx & 7;
  int lane = (idx >> 3) & 31;
  int r    = (idx >> 8) & 1;
  int nt   = (idx >> 9) & 15;
  int kk   = (idx >> 13) & 15;
  int half = r * 8 + q;
  int k = kk * 32 + r * 16 + (lane >> 4) * 8 + q;
  int n = nt * 16 + (lane & 15);
  Wz[idx] = (_Float16)W[k * DIM + n];
}

__global__ void k_zero(float* __restrict__ agg, float* __restrict__ cnt) {
  size_t stride = (size_t)gridDim.x * blockDim.x;
  size_t i0 = (size_t)blockIdx.x * blockDim.x + threadIdx.x;
  for (size_t i = i0; i < (size_t)NNODES * DIM; i += stride) agg[i] = 0.0f;
  for (size_t i = i0; i < (size_t)NNODES;       i += stride) cnt[i] = 0.0f;
}

// ---------------------------------------------------------------------------
// Cooperative async stage of the 256KB swizzled weights into LDS.
// Uses CDNA5 async copy: global_load_async_to_lds_b128, tracked by ASYNCcnt.
// blockDim.x must be 256: 256 threads * 16B * 64 iters = 262144 B.
// ---------------------------------------------------------------------------
__device__ __forceinline__ void stage_weights(const _Float16* __restrict__ Wz,
                                              _Float16* smem) {
  unsigned lbase = (unsigned)(uintptr_t)smem;               // LDS byte offset
  unsigned long long gbase = (unsigned long long)(uintptr_t)Wz;
  unsigned tid = threadIdx.x;
#pragma unroll 4
  for (unsigned it = 0; it < 64; ++it) {
    unsigned off = (it * 256u + tid) * 16u;
    unsigned laddr = lbase + off;
    unsigned long long ga = gbase + (unsigned long long)off;
    asm volatile("global_load_async_to_lds_b128 %0, %1, off"
                 :: "v"(laddr), "v"(ga) : "memory");
  }
  asm volatile("s_wait_asynccnt 0x0" ::: "memory");
  __syncthreads();
}

// ---------------------------------------------------------------------------
// A-fragment conversion. Per ISA A layout (16-bit, 16x32): lane holds row
// M=lane&15; halves 0..7 = K k0..k0+7, halves 8..15 = K k0+16..k0+23,
// k0 = (lane>>4)*8.  r0/r1 = first 8-float run, r2/r3 = second run.
// ---------------------------------------------------------------------------
__device__ __forceinline__ v16h cvt_a(v4f r0, v4f r1, v4f r2, v4f r3, float s) {
  v16h a;
  a[0]  = (_Float16)(r0[0] * s); a[1]  = (_Float16)(r0[1] * s);
  a[2]  = (_Float16)(r0[2] * s); a[3]  = (_Float16)(r0[3] * s);
  a[4]  = (_Float16)(r1[0] * s); a[5]  = (_Float16)(r1[1] * s);
  a[6]  = (_Float16)(r1[2] * s); a[7]  = (_Float16)(r1[3] * s);
  a[8]  = (_Float16)(r2[0] * s); a[9]  = (_Float16)(r2[1] * s);
  a[10] = (_Float16)(r2[2] * s); a[11] = (_Float16)(r2[3] * s);
  a[12] = (_Float16)(r3[0] * s); a[13] = (_Float16)(r3[1] * s);
  a[14] = (_Float16)(r3[2] * s); a[15] = (_Float16)(r3[3] * s);
  return a;
}

__device__ __forceinline__ void a_ptrs(const float* xrow, const float* orow,
                                       int kk, int k0,
                                       const float*& p1, const float*& p2) {
  int f1 = kk * 32 + k0;
  int f2 = f1 + 16;                 // runs never cross the 256 boundary
  p1 = (f1 < DIM) ? (xrow + f1) : (orow + (f1 - DIM));
  p2 = (f2 < DIM) ? (xrow + f2) : (orow + (f2 - DIM));
}

#define SHUF16(c0, c1) __builtin_shufflevector(c0, c1, 0,1,2,3,4,5,6,7,8,9,10,11,12,13,14,15)

// ---------------------------------------------------------------------------
// Edge MLP: weights in LDS; each wave grid-strides over 16-edge tiles,
// computing relu([x[row]|ea] @ W1 + b1) and atomically scattering into agg.
// ---------------------------------------------------------------------------
__global__ void __launch_bounds__(256) k_edge(
    const float* __restrict__ x, const long long* __restrict__ ei,
    const float* __restrict__ ea, const float* __restrict__ b1,
    const _Float16* __restrict__ W1z, float* __restrict__ agg,
    float* __restrict__ cnt) {
  extern __shared__ _Float16 smemW[];
  stage_weights(W1z, smemW);

  int lane = threadIdx.x & 31;
  int m    = lane & 15;
  int hi   = lane >> 4;
  int k0   = hi * 8;
  int n_lo = lane & 15;
  int wid0 = (int)((blockIdx.x * blockDim.x + threadIdx.x) >> 5);
  int gw   = (int)(gridDim.x * (blockDim.x >> 5));
  const int ntiles = NEDGES / 16;                   // 31250

  for (int tile = wid0; tile < ntiles; tile += gw) {   // wave-uniform
    int e0 = tile * 16;
    long long row = ei[e0 + m];
    const float* xrow  = x  + (size_t)row * DIM;
    const float* earow = ea + (size_t)(e0 + m) * DIM;

    v8f acc[16] = {};
    const float *p1, *p2;
    a_ptrs(xrow, earow, 0, k0, p1, p2);
    v4f r0 = *(const v4f*)p1, r1 = *(const v4f*)(p1 + 4);
    v4f r2 = *(const v4f*)p2, r3 = *(const v4f*)(p2 + 4);

    for (int kk = 0; kk < 16; ++kk) {
      v16h a = cvt_a(r0, r1, r2, r3, 1.0f);
      if (kk < 15) {                                 // prefetch next A tile
        a_ptrs(xrow, earow, kk + 1, k0, p1, p2);
        r0 = *(const v4f*)p1; r1 = *(const v4f*)(p1 + 4);
        r2 = *(const v4f*)p2; r3 = *(const v4f*)(p2 + 4);
      }
      int fb = (kk * 16) * 512 + lane * 8;
      v8h b0 = *(const v8h*)&smemW[fb];
      v8h b1 = *(const v8h*)&smemW[fb + 256];
#pragma unroll
      for (int nt = 0; nt < 16; ++nt) {
        v8h c0 = b0, c1 = b1;
        if (nt < 15) {                               // prefetch next B frag
          b0 = *(const v8h*)&smemW[fb + (nt + 1) * 512];
          b1 = *(const v8h*)&smemW[fb + (nt + 1) * 512 + 256];
        }
        v16h b = SHUF16(c0, c1);
        acc[nt] = __builtin_amdgcn_wmma_f32_16x16x32_f16(
            false, a, false, b, (short)0, acc[nt], false, false);
      }
    }

    // C layout: VGPR j holds row M = hi*8 + j, col N = n_lo + nt*16
    int dest[8];
#pragma unroll
    for (int j = 0; j < 8; ++j)
      dest[j] = (int)ei[(size_t)NEDGES + e0 + hi * 8 + j];

#pragma unroll
    for (int nt = 0; nt < 16; ++nt) {
      float bias = b1[nt * 16 + n_lo];
#pragma unroll
      for (int j = 0; j < 8; ++j) {
        float v = acc[nt][j] + bias;
        v = v > 0.0f ? v : 0.0f;
        unsafeAtomicAdd(&agg[(size_t)dest[j] * DIM + nt * 16 + n_lo], v);
      }
    }
    if (lane < 16) {
      int d = (int)ei[(size_t)NEDGES + e0 + lane];
      unsafeAtomicAdd(&cnt[d], 1.0f);
    }
  }
}

// ---------------------------------------------------------------------------
// Node MLP: relu([x | agg/max(cnt,1)] @ W2 + b2), weights in LDS.
// ---------------------------------------------------------------------------
__global__ void __launch_bounds__(256) k_node(
    const float* __restrict__ x, const float* __restrict__ agg,
    const float* __restrict__ cnt, const float* __restrict__ b2,
    const _Float16* __restrict__ W2z, float* __restrict__ out) {
  extern __shared__ _Float16 smemW[];
  stage_weights(W2z, smemW);

  int lane = threadIdx.x & 31;
  int m    = lane & 15;
  int hi   = lane >> 4;
  int k0   = hi * 8;
  int n_lo = lane & 15;
  int wid0 = (int)((blockIdx.x * blockDim.x + threadIdx.x) >> 5);
  int gw   = (int)(gridDim.x * (blockDim.x >> 5));
  const int ntiles = NNODES / 16;                    // 3125

  for (int tile = wid0; tile < ntiles; tile += gw) {
    int node = tile * 16 + m;
    float c   = cnt[node];
    float inv = 1.0f / (c > 1.0f ? c : 1.0f);
    const float* xrow = x   + (size_t)node * DIM;
    const float* arow = agg + (size_t)node * DIM;

    v8f acc[16] = {};
    const float *p1, *p2;
    a_ptrs(xrow, arow, 0, k0, p1, p2);
    v4f r0 = *(const v4f*)p1, r1 = *(const v4f*)(p1 + 4);
    v4f r2 = *(const v4f*)p2, r3 = *(const v4f*)(p2 + 4);

    for (int kk = 0; kk < 16; ++kk) {
      v16h a = cvt_a(r0, r1, r2, r3, (kk < 8) ? 1.0f : inv);
      if (kk < 15) {
        a_ptrs(xrow, arow, kk + 1, k0, p1, p2);
        r0 = *(const v4f*)p1; r1 = *(const v4f*)(p1 + 4);
        r2 = *(const v4f*)p2; r3 = *(const v4f*)(p2 + 4);
      }
      int fb = (kk * 16) * 512 + lane * 8;
      v8h b0 = *(const v8h*)&smemW[fb];
      v8h b1 = *(const v8h*)&smemW[fb + 256];
#pragma unroll
      for (int nt = 0; nt < 16; ++nt) {
        v8h c0 = b0, c1 = b1;
        if (nt < 15) {
          b0 = *(const v8h*)&smemW[fb + (nt + 1) * 512];
          b1 = *(const v8h*)&smemW[fb + (nt + 1) * 512 + 256];
        }
        v16h b = SHUF16(c0, c1);
        acc[nt] = __builtin_amdgcn_wmma_f32_16x16x32_f16(
            false, a, false, b, (short)0, acc[nt], false, false);
      }
    }

#pragma unroll
    for (int nt = 0; nt < 16; ++nt) {
      float bias = b2[nt * 16 + n_lo];
#pragma unroll
      for (int j = 0; j < 8; ++j) {
        float v = acc[nt][j] + bias;
        v = v > 0.0f ? v : 0.0f;
        out[(size_t)(tile * 16 + hi * 8 + j) * DIM + nt * 16 + n_lo] = v;
      }
    }
  }
}

// ---------------------------------------------------------------------------
extern "C" void kernel_launch(void* const* d_in, const int* in_sizes, int n_in,
                              void* d_out, int out_size, void* d_ws,
                              size_t ws_size, hipStream_t stream) {
  const float*     x  = (const float*)d_in[0];
  const long long* ei = (const long long*)d_in[1];   // int64 edge_index [2,E]
  const float*     ea = (const float*)d_in[2];
  // d_in[3] = u, d_in[4] = batch (unused by the node model)
  const float*     W1 = (const float*)d_in[5];
  const float*     b1 = (const float*)d_in[6];
  const float*     W2 = (const float*)d_in[7];
  const float*     b2 = (const float*)d_in[8];
  float* out = (float*)d_out;

  // workspace: W1z | W2z | agg (50000*256 f32) | cnt (50000 f32)
  char* ws = (char*)d_ws;
  _Float16* W1z = (_Float16*)(ws);
  _Float16* W2z = (_Float16*)(ws + WBYTES);
  float*    agg = (float*)   (ws + 2 * WBYTES);
  float*    cnt = (float*)   (ws + 2 * WBYTES + (size_t)NNODES * DIM * 4);

  k_swizzle<<<131072 / 256, 256, 0, stream>>>(W1, W1z);
  k_swizzle<<<131072 / 256, 256, 0, stream>>>(W2, W2z);
  k_zero<<<2048, 256, 0, stream>>>(agg, cnt);

  k_edge<<<1024, 256, WBYTES, stream>>>(x, ei, ea, b1, W1z, agg, cnt);
  k_node<<<256, 256, WBYTES, stream>>>(x, agg, cnt, b2, W2z, out);
}
